// Swin_T_43310450213371
// MI455X (gfx1250) — matrix operand order repaired and checked by reading, not compiled, not run on hardware
//
#include <hip/hip_runtime.h>

typedef __attribute__((ext_vector_type(16))) __bf16 v16bf;
typedef __attribute__((ext_vector_type(8)))  float  v8f;

// ---------------------------------------------------------------------------
// Generic GEMM: Y = X @ W^T (+bias) (+GELU) (+residual)
//   X: M x K row-major f32, W: N x K row-major f32, Y: M x N f32
//   K must be a multiple of 32 (true for every GEMM in Swin-T here).
//   flags bit0: exact GELU, bit1: add residual res[m*N+n].
// Block: 256 threads = 8 waves; tile 64(M) x 64(N) x 32(K).
// Wave grid 2(M) x 4(N); each wave: 1 B-frag, 2 A-frags, 2 WMMAs per K-step.
// OOB rows are handled by clamping the *address* (loads stay branch-free,
// garbage only lands in output rows/cols that the store-side guard drops).
// ---------------------------------------------------------------------------
__global__ __launch_bounds__(256) void gemm_wmma(
    const float* __restrict__ X, const float* __restrict__ Wt,
    const float* __restrict__ bias, const float* __restrict__ res,
    float* __restrict__ Y, int M, int N, int K, int flags)
{
    __shared__ __bf16 As[64][40];
    __shared__ __bf16 Bs[64][40];
    const int tid  = threadIdx.x;
    const int lane = tid & 31, wave = tid >> 5;
    const int wm = wave >> 2, wn = wave & 3;          // 2 x 4 wave grid
    const int m0 = blockIdx.y * 64, n0 = blockIdx.x * 64;

    // Per-thread staging coords: 64 rows x 8 float4-quads, 2 rounds.
    const int lr = tid >> 3;                          // 0..31
    const int lq = (tid & 7) * 4;                     // 0,4,...,28

    v8f acc[2] = {v8f{}, v8f{}};

    for (int k0 = 0; k0 < K; k0 += 32) {
        #pragma unroll
        for (int half = 0; half < 2; ++half) {
            const int r  = lr + half * 32;
            int gm = m0 + r; gm = (gm < M) ? gm : (M - 1);
            int gn = n0 + r; gn = (gn < N) ? gn : (N - 1);
            const float4 va = *(const float4*)(X  + (size_t)gm * K + k0 + lq);
            const float4 vb = *(const float4*)(Wt + (size_t)gn * K + k0 + lq);
            As[r][lq + 0] = (__bf16)va.x; As[r][lq + 1] = (__bf16)va.y;
            As[r][lq + 2] = (__bf16)va.z; As[r][lq + 3] = (__bf16)va.w;
            Bs[r][lq + 0] = (__bf16)vb.x; Bs[r][lq + 1] = (__bf16)vb.y;
            Bs[r][lq + 2] = (__bf16)vb.z; Bs[r][lq + 3] = (__bf16)vb.w;
        }
        __syncthreads();

        v16bf b;
        const int ncol = wn * 16 + (lane & 15);
        const int kb2  = (lane >> 4) << 3;            // 0 or 8
        #pragma unroll
        for (int j = 0; j < 16; ++j) b[j] = Bs[ncol][(j < 8) ? (kb2 + j) : (16 + kb2 + (j - 8))];

        v16bf a0, a1;
        const int mrow = wm * 32 + (lane & 15);
        const int kb   = (lane >> 4) << 3;            // 0 or 8
        #pragma unroll
        for (int j = 0; j < 16; ++j) {
            const int kk = (j < 8) ? (kb + j) : (16 + kb + (j - 8));
            a0[j] = As[mrow][kk];
            a1[j] = As[mrow + 16][kk];
        }

        acc[0] = __builtin_amdgcn_wmma_f32_16x16x32_bf16(
            false, a0, false, b, (short)0, acc[0], false, false);
        acc[1] = __builtin_amdgcn_wmma_f32_16x16x32_bf16(
            false, a1, false, b, (short)0, acc[1], false, false);
        __syncthreads();
    }

    const int n = n0 + wn * 16 + (lane & 15);
    if (n < N) {
        const float bv = bias ? bias[n] : 0.0f;
        #pragma unroll
        for (int t = 0; t < 2; ++t) {
            const int mbase = m0 + wm * 32 + t * 16 + ((lane >> 4) << 3);
            #pragma unroll
            for (int r = 0; r < 8; ++r) {
                const int m = mbase + r;
                if (m < M) {
                    float v = acc[t][r] + bv;
                    if (flags & 1) v = 0.5f * v * (1.0f + erff(v * 0.70710678118654752f));
                    if (flags & 2) v += res[(size_t)m * N + n];
                    Y[(size_t)m * N + n] = v;
                }
            }
        }
    }
}

// ---------------------------------------------------------------------------
// Fused window attention: one block (128 thr = 4 waves) per (batch, win, head).
// qkv: (B*H*W, 3C) token-major (rolled gather via index math), head_dim = 32.
// out: (B*H*W, C) with inverse roll applied by scatter.
// ---------------------------------------------------------------------------
__global__ __launch_bounds__(128) void win_attn(
    const float* __restrict__ qkv, const float* __restrict__ rpb,
    float* __restrict__ out, int H, int C, int nh, int shift)
{
    const int ws = 7, T = 49;
    const int nW = H / ws;
    int idx = blockIdx.x;
    const int h  = idx % nh;  idx /= nh;
    const int wwi = idx % nW; idx /= nW;
    const int whi = idx % nW; idx /= nW;
    const int b  = idx;
    const int tid = threadIdx.x, lane = tid & 31, wave = tid >> 5;
    const int C3 = 3 * C;
    const float scale = 0.17677669529663687f;  // 1/sqrt(32)

    __shared__ __bf16 Qs[64][40];
    __shared__ __bf16 Ks[64][40];
    __shared__ __bf16 Vt[32][72];   // transposed: [d][token]
    __shared__ float  S [64][65];
    __shared__ __bf16 Ps[64][72];

    // Load Q,K,V (rolled gather) with float4 loads; zero-pad tokens 49..63.
    {
        const int t = tid >> 1, dp = (tid & 1) * 16;
        const int r = t / ws, c = t % ws;
        const int gr = (whi * ws + r + shift) % H;
        const int gc = (wwi * ws + c + shift) % H;
        const size_t base = ((size_t)(b * H + gr) * H + gc) * C3 + (size_t)h * 32 + dp;
        if (t < T) {
            #pragma unroll
            for (int d4 = 0; d4 < 16; d4 += 4) {
                const float4 q = *(const float4*)(qkv + base + d4);
                const float4 k = *(const float4*)(qkv + base + C + d4);
                const float4 v = *(const float4*)(qkv + base + 2 * C + d4);
                const int d = dp + d4;
                Qs[t][d + 0] = (__bf16)(q.x * scale);
                Qs[t][d + 1] = (__bf16)(q.y * scale);
                Qs[t][d + 2] = (__bf16)(q.z * scale);
                Qs[t][d + 3] = (__bf16)(q.w * scale);
                Ks[t][d + 0] = (__bf16)k.x; Ks[t][d + 1] = (__bf16)k.y;
                Ks[t][d + 2] = (__bf16)k.z; Ks[t][d + 3] = (__bf16)k.w;
                Vt[d + 0][t] = (__bf16)v.x; Vt[d + 1][t] = (__bf16)v.y;
                Vt[d + 2][t] = (__bf16)v.z; Vt[d + 3][t] = (__bf16)v.w;
            }
        } else {
            #pragma unroll
            for (int d4 = 0; d4 < 16; ++d4) {
                const int d = dp + d4;
                Qs[t][d] = (__bf16)0.f;
                Ks[t][d] = (__bf16)0.f;
                Vt[d][t] = (__bf16)0.f;
            }
        }
    }
    __syncthreads();

    // S = Q K^T  (64x64 padded, K=32): 16 tiles over 4 waves.
    for (int tile = wave; tile < 16; tile += 4) {
        const int tm = tile >> 2, tn = tile & 3;
        v16bf a, bf;
        const int mrow = tm * 16 + (lane & 15);
        const int kb   = (lane >> 4) << 3;
        #pragma unroll
        for (int j = 0; j < 16; ++j) {
            const int kk = (j < 8) ? (kb + j) : (16 + kb + (j - 8));
            a[j] = Qs[mrow][kk];
        }
        const int ncol = tn * 16 + (lane & 15);
        const int kb2  = (lane >> 4) << 3;
        #pragma unroll
        for (int j = 0; j < 16; ++j) bf[j] = Ks[ncol][(j < 8) ? (kb2 + j) : (16 + kb2 + (j - 8))];
        v8f c = {};
        c = __builtin_amdgcn_wmma_f32_16x16x32_bf16(
            false, a, false, bf, (short)0, c, false, false);
        const int sn  = tn * 16 + (lane & 15);
        const int smb = tm * 16 + ((lane >> 4) << 3);
        #pragma unroll
        for (int r = 0; r < 8; ++r) S[smb + r][sn] = c[r];
    }
    __syncthreads();

    // rpb + shift-mask + softmax, one row per thread (tid < 64).
    if (tid < 64) {
        const int i = tid;
        if (i < T) {
            const int yi = i / ws, xi = i % ws;
            int gi = 0;
            if (shift) {
                const int ry = whi * ws + yi, rx = wwi * ws + xi;
                const int cy = (ry >= H - shift) ? 2 : ((ry >= H - ws) ? 1 : 0);
                const int cx = (rx >= H - shift) ? 2 : ((rx >= H - ws) ? 1 : 0);
                gi = 3 * cy + cx;
            }
            float mx = -1e30f;
            for (int j = 0; j < T; ++j) {
                const int yj = j / ws, xj = j % ws;
                const int ridx = (yi - yj + 6) * 13 + (xi - xj + 6);
                float v = S[i][j] + rpb[ridx * nh + h];
                if (shift) {
                    const int ry = whi * ws + yj, rx = wwi * ws + xj;
                    const int cy = (ry >= H - shift) ? 2 : ((ry >= H - ws) ? 1 : 0);
                    const int cx = (rx >= H - shift) ? 2 : ((rx >= H - ws) ? 1 : 0);
                    if (3 * cy + cx != gi) v -= 100.0f;
                }
                S[i][j] = v;
                mx = fmaxf(mx, v);
            }
            float sum = 0.f;
            for (int j = 0; j < T; ++j) {
                const float e = __expf(S[i][j] - mx);
                S[i][j] = e; sum += e;
            }
            const float inv = 1.0f / sum;
            for (int j = 0; j < T;  ++j) Ps[i][j] = (__bf16)(S[i][j] * inv);
            for (int j = T; j < 64; ++j) Ps[i][j] = (__bf16)0.f;
        } else {
            for (int j = 0; j < 64; ++j) Ps[i][j] = (__bf16)0.f;
        }
    }
    __syncthreads();

    // O = P @ V  (64x32, K=64): 8 tiles over 4 waves, 2 K-steps each.
    for (int tile = wave; tile < 8; tile += 4) {
        const int tm = tile >> 1, tn = tile & 1;
        v8f c = {};
        #pragma unroll
        for (int ks = 0; ks < 64; ks += 32) {
            v16bf a, bf;
            const int mrow = tm * 16 + (lane & 15);
            const int kb   = (lane >> 4) << 3;
            #pragma unroll
            for (int j = 0; j < 16; ++j) {
                const int kk = ks + ((j < 8) ? (kb + j) : (16 + kb + (j - 8)));
                a[j] = Ps[mrow][kk];
            }
            const int ncol = tn * 16 + (lane & 15);
            const int kb2  = ks + ((lane >> 4) << 3);
            #pragma unroll
            for (int j = 0; j < 16; ++j) bf[j] = Vt[ncol][(j < 8) ? (kb2 + j) : (16 + kb2 + (j - 8))];
            c = __builtin_amdgcn_wmma_f32_16x16x32_bf16(
                false, a, false, bf, (short)0, c, false, false);
        }
        const int d  = tn * 16 + (lane & 15);
        const int mb = tm * 16 + ((lane >> 4) << 3);
        #pragma unroll
        for (int r = 0; r < 8; ++r) {
            const int i = mb + r;
            if (i < T) {
                const int rr = i / ws, cc = i % ws;
                const int gr = (whi * ws + rr + shift) % H;
                const int gc = (wwi * ws + cc + shift) % H;
                out[((size_t)(b * H + gr) * H + gc) * C + h * 32 + d] = c[r];
            }
        }
    }
}

// ---------------------------------------------------------------------------
// LayerNorm: one wave per row (8 rows per 256-thread block).
// ---------------------------------------------------------------------------
__global__ __launch_bounds__(256) void layernorm_k(
    const float* __restrict__ X, const float* __restrict__ g,
    const float* __restrict__ bt, float* __restrict__ Y, int rows, int C)
{
    const int wave = threadIdx.x >> 5, lane = threadIdx.x & 31;
    const int row = blockIdx.x * 8 + wave;
    if (row >= rows) return;
    const float* xr = X + (size_t)row * C;
    float s = 0.f;
    for (int c = lane; c < C; c += 32) s += xr[c];
    #pragma unroll
    for (int off = 16; off; off >>= 1) s += __shfl_xor(s, off, 32);
    const float mu = s / C;
    float v = 0.f;
    for (int c = lane; c < C; c += 32) { const float d = xr[c] - mu; v += d * d; }
    #pragma unroll
    for (int off = 16; off; off >>= 1) v += __shfl_xor(v, off, 32);
    const float rstd = rsqrtf(v / C + 1e-5f);
    float* yr = Y + (size_t)row * C;
    for (int c = lane; c < C; c += 32) yr[c] = (xr[c] - mu) * rstd * g[c] + bt[c];
}

// Patch embedding: 4x4/4 conv as direct per-output dot (0.9 GFLOP, negligible).
__global__ void patch_embed(const float* __restrict__ x, const float* __restrict__ w,
                            const float* __restrict__ bias, float* __restrict__ y)
{
    const int total = 32 * 56 * 56 * 96;
    const int i = blockIdx.x * blockDim.x + threadIdx.x;
    if (i >= total) return;
    const int co = i % 96; int t = i / 96;
    const int ox = t % 56; t /= 56;
    const int oy = t % 56; const int b = t / 56;
    float s = bias[co];
    for (int ci = 0; ci < 3; ++ci)
        for (int ky = 0; ky < 4; ++ky)
            for (int kx = 0; kx < 4; ++kx)
                s += x[(((size_t)b * 3 + ci) * 224 + (oy * 4 + ky)) * 224 + (ox * 4 + kx)]
                   * w[((co * 3 + ci) * 4 + ky) * 4 + kx];
    y[i] = s;
}

// Patch-merge gather: (B,H,H,C) -> (B,H/2,H/2,4C), quadrants (0,0),(1,0),(0,1),(1,1).
__global__ void merge_gather(const float* __restrict__ X, float* __restrict__ Y,
                             int H, int C)
{
    const int Ho = H / 2;
    const size_t total = (size_t)32 * Ho * Ho * 4 * C;
    const size_t i = (size_t)blockIdx.x * blockDim.x + threadIdx.x;
    if (i >= total) return;
    const int e = (int)(i % (4 * C));
    const int ch = e % C, q = e / C;
    size_t t = i / (4 * C);
    const int ox = (int)(t % Ho); t /= Ho;
    const int oy = (int)(t % Ho); const int b = (int)(t / Ho);
    const int dy = q & 1, dx = q >> 1;
    Y[i] = X[(((size_t)b * H + (2 * oy + dy)) * H + (2 * ox + dx)) * C + ch];
}

__global__ void mean_pool(const float* __restrict__ X, float* __restrict__ Y,
                          int HW, int C)
{
    const int i = blockIdx.x * blockDim.x + threadIdx.x;
    if (i >= 32 * C) return;
    const int c = i % C, b = i / C;
    float s = 0.f;
    for (int t = 0; t < HW; ++t) s += X[((size_t)b * HW + t) * C + c];
    Y[i] = s / HW;
}

__global__ void head_k(const float* __restrict__ X, const float* __restrict__ W,
                       const float* __restrict__ bias, float* __restrict__ Y,
                       int C, int N)
{
    const int i = blockIdx.x * blockDim.x + threadIdx.x;
    if (i >= 32 * N) return;
    const int n = i % N, b = i / N;
    float s = bias[n];
    for (int c = 0; c < C; ++c) s += X[b * C + c] * W[n * C + c];
    Y[i] = s;
}

// ---------------------------------------------------------------------------
// Host orchestration
// ---------------------------------------------------------------------------
namespace {
struct BlockP {
    const float *ln1_g, *ln1_b, *qkv_w, *qkv_b, *rpb, *proj_w, *proj_b,
                *ln2_g, *ln2_b, *fc1_w, *fc1_b, *fc2_w, *fc2_b;
};
struct MergeP { const float *ln_g, *ln_b, *red_w; };
struct Net {
    const float *x, *patch_w, *patch_b, *patch_ln_g, *patch_ln_b;
    const float *ln_g, *ln_b, *head_w, *head_b;
    BlockP blocks[4][6];
    MergeP merge[3];
};
const int kDepths[4] = {2, 2, 6, 2};
const int kDims[4]   = {96, 192, 384, 768};
const int kHeads[4]  = {3, 6, 12, 24};
const int kHs[4]     = {56, 28, 14, 7};

void fill_params(void* const* d_in, const int* in_sizes, Net& net) {
    int i = 0;
    auto nx = [&]() { return (const float*)d_in[i++]; };
    const bool insertion = (in_sizes[0] == 32 * 3 * 224 * 224);
    if (insertion) {  // python dict insertion order, recursive
        net.x = nx(); net.patch_w = nx(); net.patch_b = nx();
        net.patch_ln_g = nx(); net.patch_ln_b = nx();
        net.ln_g = nx(); net.ln_b = nx(); net.head_w = nx(); net.head_b = nx();
        for (int s = 0; s < 4; ++s) {
            for (int bk = 0; bk < kDepths[s]; ++bk) {
                BlockP& p = net.blocks[s][bk];
                p.ln1_g = nx(); p.ln1_b = nx(); p.qkv_w = nx(); p.qkv_b = nx();
                p.rpb = nx(); p.proj_w = nx(); p.proj_b = nx();
                p.ln2_g = nx(); p.ln2_b = nx();
                p.fc1_w = nx(); p.fc1_b = nx(); p.fc2_w = nx(); p.fc2_b = nx();
            }
            if (s < 3) {
                net.merge[s].ln_g = nx(); net.merge[s].ln_b = nx();
                net.merge[s].red_w = nx();
            }
        }
    } else {  // jax tree_leaves: sorted dict keys, params then x
        net.head_b = nx(); net.head_w = nx(); net.ln_b = nx(); net.ln_g = nx();
        net.patch_b = nx(); net.patch_ln_b = nx(); net.patch_ln_g = nx();
        net.patch_w = nx();
        for (int s = 0; s < 4; ++s) {
            for (int bk = 0; bk < kDepths[s]; ++bk) {
                BlockP& p = net.blocks[s][bk];
                p.fc1_b = nx(); p.fc1_w = nx(); p.fc2_b = nx(); p.fc2_w = nx();
                p.ln1_b = nx(); p.ln1_g = nx(); p.ln2_b = nx(); p.ln2_g = nx();
                p.proj_b = nx(); p.proj_w = nx(); p.qkv_b = nx(); p.qkv_w = nx();
                p.rpb = nx();
            }
            if (s < 3) {
                net.merge[s].ln_b = nx(); net.merge[s].ln_g = nx();
                net.merge[s].red_w = nx();
            }
        }
        net.x = nx();
    }
}
}  // namespace

extern "C" void kernel_launch(void* const* d_in, const int* in_sizes, int n_in,
                              void* d_out, int out_size, void* d_ws, size_t ws_size,
                              hipStream_t stream) {
    (void)n_in; (void)out_size; (void)ws_size;
    Net net;
    fill_params(d_in, in_sizes, net);

    float* b0 = (float*)d_ws;          // current x / residual  (<= 9.7M f32)
    float* b1 = b0 + 9700000;          // temp                   (<= 9.7M f32)
    float* b2 = b1 + 9700000;          // qkv / fc1-out          (<= 38.6M f32)
    float* b3 = b2 + 38600000;         // temp                   (<= 9.7M f32)

    // Patch embedding + LN
    const int tok0 = 32 * 56 * 56;
    patch_embed<<<(tok0 * 96 + 255) / 256, 256, 0, stream>>>(
        net.x, net.patch_w, net.patch_b, b1);
    layernorm_k<<<(tok0 + 7) / 8, 256, 0, stream>>>(
        b1, net.patch_ln_g, net.patch_ln_b, b0, tok0, 96);

    for (int s = 0; s < 4; ++s) {
        const int C = kDims[s], H = kHs[s], nh = kHeads[s];
        const int tokens = 32 * H * H;
        const int nW = H / 7;
        for (int bk = 0; bk < kDepths[s]; ++bk) {
            const int shift = (bk & 1) ? 3 : 0;
            const BlockP& p = net.blocks[s][bk];
            layernorm_k<<<(tokens + 7) / 8, 256, 0, stream>>>(
                b0, p.ln1_g, p.ln1_b, b1, tokens, C);
            gemm_wmma<<<dim3((3 * C + 63) / 64, (tokens + 63) / 64), 256, 0, stream>>>(
                b1, p.qkv_w, p.qkv_b, nullptr, b2, tokens, 3 * C, C, 0);
            win_attn<<<32 * nW * nW * nh, 128, 0, stream>>>(
                b2, p.rpb, b3, H, C, nh, shift);
            gemm_wmma<<<dim3((C + 63) / 64, (tokens + 63) / 64), 256, 0, stream>>>(
                b3, p.proj_w, p.proj_b, b0, b1, tokens, C, C, 2);      // x1 -> b1
            layernorm_k<<<(tokens + 7) / 8, 256, 0, stream>>>(
                b1, p.ln2_g, p.ln2_b, b3, tokens, C);
            gemm_wmma<<<dim3((4 * C + 63) / 64, (tokens + 63) / 64), 256, 0, stream>>>(
                b3, p.fc1_w, p.fc1_b, nullptr, b2, tokens, 4 * C, C, 1);  // GELU
            gemm_wmma<<<dim3((C + 63) / 64, (tokens + 63) / 64), 256, 0, stream>>>(
                b2, p.fc2_w, p.fc2_b, b1, b0, tokens, C, 4 * C, 2);    // x2 -> b0
        }
        if (s < 3) {
            const int Ho = H / 2, tok2 = 32 * Ho * Ho;
            const size_t tot = (size_t)tok2 * 4 * C;
            merge_gather<<<(unsigned)((tot + 255) / 256), 256, 0, stream>>>(b0, b2, H, C);
            layernorm_k<<<(tok2 + 7) / 8, 256, 0, stream>>>(
                b2, net.merge[s].ln_g, net.merge[s].ln_b, b1, tok2, 4 * C);
            gemm_wmma<<<dim3((2 * C + 63) / 64, (tok2 + 63) / 64), 256, 0, stream>>>(
                b1, net.merge[s].red_w, nullptr, nullptr, b0, tok2, 2 * C, 4 * C, 0);
        }
    }

    // Final LN, global average pool, head
    const int tokf = 32 * 49;
    layernorm_k<<<(tokf + 7) / 8, 256, 0, stream>>>(
        b0, net.ln_g, net.ln_b, b1, tokf, 768);
    mean_pool<<<(32 * 768 + 255) / 256, 256, 0, stream>>>(b1, b3, 49, 768);
    head_k<<<1, 256, 0, stream>>>(b3, net.head_w, net.head_b, (float*)d_out, 768, 3);
}
